// SparseTwoCliqueAttentionLayer_65403761983981
// MI455X (gfx1250) — matrix operand order repaired and checked by reading, not compiled, not run on hardware
//
#include <hip/hip_runtime.h>
#include <hip/hip_bf16.h>

typedef __attribute__((ext_vector_type(16))) _Float16 v16h;
typedef __attribute__((ext_vector_type(8)))  _Float16 v8h;
typedef __attribute__((ext_vector_type(8)))  float    v8f;

#define HIDDEN 128
#define WT_STRIDE 136   // 128 halves + 8-half pad: 272B row = 68 dwords = 4-bank
                        // stride -> conflict-free b128 B-fragment reads, 16B aligned

// ---------------------------------------------------------------------------
// Kernel 1: Q/K/V projection.
//   grid = ( ceil(rowTiles/8), 3 )   blockIdx.y selects projection (Q/K/V)
//   block = 256 threads = 8 waves; each wave owns one 16-row tile of x,
//   holds all four K-chunk A fragments in registers (float4 loads, cvt once),
//   and sweeps the 8 column tiles -> 32 v_wmma_f32_16x16x32_f16 per wave.
//   The 128x128 weight is staged once per block into LDS *transposed* as f16
//   (WldsT[col][k], padded), so every B fragment is 16 consecutive halves per
//   lane: two conflict-free ds_load_b128 per fragment.
//   Bias is fused by pre-loading the C fragment (C/D layout: N = lane%16).
// ---------------------------------------------------------------------------
__global__ __launch_bounds__(256)
void qkv_wmma_kernel(const float* __restrict__ x,
                     const float* __restrict__ Qw, const float* __restrict__ Qb,
                     const float* __restrict__ Kw, const float* __restrict__ Kb,
                     const float* __restrict__ Vw, const float* __restrict__ Vb,
                     float* __restrict__ qkv,      // [3][nN][128]
                     int nN, int rowTiles)
{
    __shared__ _Float16 WldsT[HIDDEN * WT_STRIDE];   // ~34 KB of 320 KB WGP LDS

    const int p = blockIdx.y;                      // 0=Q, 1=K, 2=V
    const float* W  = (p == 0) ? Qw : (p == 1) ? Kw : Vw;
    const float* Bb = (p == 0) ? Qb : (p == 1) ? Kb : Vb;

    // cooperative transpose-stage: coalesced float4 global reads, f32->f16,
    // scatter to col-major LDS. 16 float4 loads + 64 stores per thread, once.
    for (int base = threadIdx.x * 4; base < HIDDEN * HIDDEN; base += 256 * 4) {
        const float4 w4 = *(const float4*)(W + base);
        const int k   = base >> 7;                 // row of W  (K index)
        const int col = base & 127;                // col of W  (N index)
        WldsT[(col + 0) * WT_STRIDE + k] = (_Float16)w4.x;
        WldsT[(col + 1) * WT_STRIDE + k] = (_Float16)w4.y;
        WldsT[(col + 2) * WT_STRIDE + k] = (_Float16)w4.z;
        WldsT[(col + 3) * WT_STRIDE + k] = (_Float16)w4.w;
    }
    __syncthreads();

    const int wave = threadIdx.x >> 5;
    const int lane = threadIdx.x & 31;
    const int rt   = blockIdx.x * 8 + wave;        // row tile
    if (rt >= rowTiles) return;

    const int halfSel = lane >> 4;                 // 0: lanes 0-15, 1: 16-31
    const int l16     = lane & 15;

    const int row  = rt * 16 + l16;                // A fragment: M = lane%16
    const int rowc = row < nN ? row : nN - 1;      // clamp (loads only)

    const int kbaseA = halfSel * 16;               // A: lanes 16-31 -> K=16..31
    const int kbaseB = halfSel * 16;               // B: lanes 16-31 -> K=16..31

    // ---- load all four 16x32 A fragments once (float4 vector loads) ----
    v16h afrag[4];
#pragma unroll
    for (int kb = 0; kb < 4; ++kb) {
        const float* xa = x + (size_t)rowc * HIDDEN + kb * 32 + kbaseA;
        const float4 lo0 = *(const float4*)(xa);
        const float4 lo1 = *(const float4*)(xa + 4);
        const float4 hi0 = *(const float4*)(xa + 8);
        const float4 hi1 = *(const float4*)(xa + 12);
        v16h a;
        a[0]  = (_Float16)lo0.x; a[1]  = (_Float16)lo0.y;
        a[2]  = (_Float16)lo0.z; a[3]  = (_Float16)lo0.w;
        a[4]  = (_Float16)lo1.x; a[5]  = (_Float16)lo1.y;
        a[6]  = (_Float16)lo1.z; a[7]  = (_Float16)lo1.w;
        a[8]  = (_Float16)hi0.x; a[9]  = (_Float16)hi0.y;
        a[10] = (_Float16)hi0.z; a[11] = (_Float16)hi0.w;
        a[12] = (_Float16)hi1.x; a[13] = (_Float16)hi1.y;
        a[14] = (_Float16)hi1.z; a[15] = (_Float16)hi1.w;
        afrag[kb] = a;
    }

    float* outBase = qkv + (size_t)p * nN * HIDDEN + (size_t)rt * 16 * HIDDEN;

    // ---- sweep the 8 column tiles ----
#pragma unroll
    for (int ct = 0; ct < 8; ++ct) {
        const int col = ct * 16 + l16;             // B/C/D: N = lane%16

        // C fragment = bias broadcast (bias depends only on N)
        const float bias = Bb[col];
        v8f acc;
#pragma unroll
        for (int r = 0; r < 8; ++r) acc[r] = bias;

#pragma unroll
        for (int kb = 0; kb < 4; ++kb) {
            // B fragment 32x16 f16: halves e -> K = kb*32 + kbaseB + e.
            // Col-major LDS => 16 consecutive halves: two aligned b128 reads.
            const _Float16* wb = WldsT + col * WT_STRIDE + kb * 32 + kbaseB;
            const v8h blo = *(const v8h*)(wb);
            const v8h bhi = *(const v8h*)(wb + 8);
            v16h b;
#pragma unroll
            for (int e = 0; e < 8; ++e) { b[e] = blo[e]; b[8 + e] = bhi[e]; }

            acc = __builtin_amdgcn_wmma_f32_16x16x32_f16(
                /*neg_a=*/false, afrag[kb], /*neg_b=*/false, b,
                /*c_mod=*/(short)0, acc, /*reuse_a=*/false, /*reuse_b=*/false);
        }

        // D fragment store: VGPR r -> (M = r + 8*halfSel, N = l16)
#pragma unroll
        for (int r = 0; r < 8; ++r) {
            const int m = r + halfSel * 8;
            if (rt * 16 + m < nN)
                outBase[(size_t)m * HIDDEN + ct * 16 + l16] = acc[r];
        }
    }
}

// ---------------------------------------------------------------------------
// Kernel 2: one wave per clique. 32 lanes x float4 = 128 dims.
// pre = (6-term symmetric contraction summed over all 128 dims) / 24
// diagA2[t] = exp(pre).  Gathers hit L2 (q/k/v = 154MB < 192MB L2).
// ---------------------------------------------------------------------------
__global__ __launch_bounds__(256)
void clique_score_kernel(const float* __restrict__ q,
                         const float* __restrict__ k,
                         const float* __restrict__ v,
                         const int* __restrict__ tci,
                         float* __restrict__ diagA2, int nC)
{
    const int wave = threadIdx.x >> 5;
    const int lane = threadIdx.x & 31;
    const int t = blockIdx.x * 8 + wave;
    if (t >= nC) return;

    const int i  = tci[t];
    const int j  = tci[nC + t];
    const int kc = tci[2 * nC + t];
    const int d0 = lane * 4;

    const float4 q_i = *(const float4*)(q + (size_t)i  * HIDDEN + d0);
    const float4 q_j = *(const float4*)(q + (size_t)j  * HIDDEN + d0);
    const float4 q_k = *(const float4*)(q + (size_t)kc * HIDDEN + d0);
    const float4 k_i = *(const float4*)(k + (size_t)i  * HIDDEN + d0);
    const float4 k_j = *(const float4*)(k + (size_t)j  * HIDDEN + d0);
    const float4 k_k = *(const float4*)(k + (size_t)kc * HIDDEN + d0);
    const float4 v_i = *(const float4*)(v + (size_t)i  * HIDDEN + d0);
    const float4 v_j = *(const float4*)(v + (size_t)j  * HIDDEN + d0);
    const float4 v_k = *(const float4*)(v + (size_t)kc * HIDDEN + d0);

#define SIX_TERM(c) (q_i.c * k_j.c * v_k.c + q_j.c * k_k.c * v_i.c + \
                     q_k.c * k_i.c * v_j.c + q_j.c * k_i.c * v_k.c + \
                     q_k.c * k_j.c * v_i.c + q_i.c * k_k.c * v_j.c)
    float s = SIX_TERM(x) + SIX_TERM(y) + SIX_TERM(z) + SIX_TERM(w);
#undef SIX_TERM

    // wave32 reduction
#pragma unroll
    for (int off = 16; off > 0; off >>= 1)
        s += __shfl_xor(s, off, 32);

    if (lane == 0)
        diagA2[t] = __expf(s * (1.0f / 24.0f));
}

// ---------------------------------------------------------------------------
// Kernel 3: clique -> edge scatter (hardware f32 atomics, resolve in L2)
// ---------------------------------------------------------------------------
__global__ __launch_bounds__(256)
void clique_to_edge_kernel(const float* __restrict__ diagA2,
                           const int* __restrict__ d1tgt,
                           float* __restrict__ diagA1, int m3)
{
    const int m = blockIdx.x * blockDim.x + threadIdx.x;
    if (m >= m3) return;
    unsafeAtomicAdd(diagA1 + d1tgt[m], diagA2[m / 3]);
}

// ---------------------------------------------------------------------------
// Kernel 4: edge -> node scatter
// ---------------------------------------------------------------------------
__global__ __launch_bounds__(256)
void edge_to_node_kernel(const float* __restrict__ diagA1,
                         const int* __restrict__ d0tgt,
                         float* __restrict__ diagA0, int m2)
{
    const int m = blockIdx.x * blockDim.x + threadIdx.x;
    if (m >= m2) return;
    unsafeAtomicAdd(diagA0 + d0tgt[m], diagA1[m >> 1]);
}

// ---------------------------------------------------------------------------
extern "C" void kernel_launch(void* const* d_in, const int* in_sizes, int n_in,
                              void* d_out, int out_size, void* d_ws, size_t ws_size,
                              hipStream_t stream)
{
    const float* x        = (const float*)d_in[0];
    const int*   d0_index = (const int*)  d_in[2];
    const int*   tci      = (const int*)  d_in[3];
    const int*   d1_index = (const int*)  d_in[4];
    const float* Qw       = (const float*)d_in[5];
    const float* Qb       = (const float*)d_in[6];
    const float* Kw       = (const float*)d_in[7];
    const float* Kb       = (const float*)d_in[8];
    const float* Vw       = (const float*)d_in[9];
    const float* Vb       = (const float*)d_in[10];

    const int nN = in_sizes[0] / HIDDEN;   // 100000
    const int nE = in_sizes[1] / 2;        // 1600000
    const int nC = in_sizes[3] / 3;        // 250000

    float* qkv = (float*)d_ws;             // 3 * nN * 128 floats (153.6 MB)
    float* out    = (float*)d_out;
    float* diagA0 = out;                   // [nN]
    float* diagA1 = out + nN;              // [nE]
    float* diagA2 = out + nN + nE;         // [nC]

    // 1) Q/K/V projection via WMMA (blockIdx.y = projection)
    const int rowTiles = (nN + 15) / 16;
    dim3 gemmGrid((rowTiles + 7) / 8, 3);
    qkv_wmma_kernel<<<gemmGrid, 256, 0, stream>>>(
        x, Qw, Qb, Kw, Kb, Vw, Vb, qkv, nN, rowTiles);

    // 2) zero the atomic targets (d_out is not re-poisoned between replays)
    hipMemsetAsync(d_out, 0, (size_t)(nN + nE) * sizeof(float), stream);

    // 3) clique scores (one wave per clique)
    clique_score_kernel<<<(nC + 7) / 8, 256, 0, stream>>>(
        qkv, qkv + (size_t)nN * HIDDEN, qkv + (size_t)2 * nN * HIDDEN,
        tci, diagA2, nC);

    // 4) clique -> edge
    const int m3 = 3 * nC;
    clique_to_edge_kernel<<<(m3 + 255) / 256, 256, 0, stream>>>(
        diagA2, d1_index + m3, diagA1, m3);

    // 5) edge -> node
    const int m2 = 2 * nE;
    edge_to_node_kernel<<<(m2 + 255) / 256, 256, 0, stream>>>(
        diagA1, d0_index + m2, diagA0, m2);
}
